// MultiExpertMoELayer_62380105007317
// MI455X (gfx1250) — matrix operand order repaired and checked by reading, not compiled, not run on hardware
//
#include <hip/hip_runtime.h>
#include <hip/hip_bf16.h>

// ---- model constants (from reference) ----
#define NUM_OPS   8
#define N_EXPERTS 2
#define D_MODEL   1024
#define D_FF      4096
#define BATCH     4
#define SEQ       2048
#define TOKENS    (BATCH * SEQ)   // 8192

typedef __attribute__((ext_vector_type(16))) __bf16 bf16x16;
typedef __attribute__((ext_vector_type(8)))  __bf16 bf16x8;
typedef __attribute__((ext_vector_type(4)))  __bf16 bf16x4;
typedef __attribute__((ext_vector_type(8)))  float  floatx8;
typedef __attribute__((ext_vector_type(4)))  int    i32x4;

// ---- gfx1250 async memory->LDS path (probe via __has_builtin) ----
#if defined(__HIP_DEVICE_COMPILE__) && \
    __has_builtin(__builtin_amdgcn_global_load_async_to_lds_b128) && \
    __has_builtin(__builtin_amdgcn_s_wait_asynccnt)
#define USE_ASYNC_LDS 1
#else
#define USE_ASYNC_LDS 0
#endif

#if USE_ASYNC_LDS
// builtin signature (from hipcc diagnostic): (v4i AS1*, v4i AS3*, imm offset, imm cpol)
typedef __attribute__((address_space(1))) i32x4* gbl_v4i_p;
typedef __attribute__((address_space(3))) i32x4* lds_v4i_p;
__device__ __forceinline__ void async_b128(const void* g, void* l) {
    // per-lane 16B copy, global -> LDS, tracked by ASYNCcnt
    __builtin_amdgcn_global_load_async_to_lds_b128((gbl_v4i_p)g, (lds_v4i_p)l, 0, 0);
}
#endif

// ------------------------------------------------------------------
// Kernel 1: device-side routing (argmax over first token's op band)
// ------------------------------------------------------------------
__global__ void moe_argmax_kernel(const float* __restrict__ x, int* __restrict__ op) {
    if (threadIdx.x == 0 && blockIdx.x == 0) {
        float best = x[0];
        int bi = 0;
        #pragma unroll
        for (int i = 1; i < NUM_OPS; ++i) {
            float v = x[i];
            if (v > best) { best = v; bi = i; }   // first-max tie-break like argmax
        }
        *op = bi;
    }
}

// ------------------------------------------------------------------
// Kernel 2: fp32 -> bf16 elementwise convert (activations)
// ------------------------------------------------------------------
__global__ __launch_bounds__(256) void moe_cvt_bf16_kernel(const float* __restrict__ in,
                                                           __bf16* __restrict__ out, int n4) {
    int i = blockIdx.x * blockDim.x + threadIdx.x;
    if (i < n4) {
        const float4 v = *(const float4*)(in + (size_t)i * 4);
        bf16x4 o;
        o[0] = (__bf16)v.x; o[1] = (__bf16)v.y; o[2] = (__bf16)v.z; o[3] = (__bf16)v.w;
        *(bf16x4*)(out + (size_t)i * 4) = o;
    }
}

// ------------------------------------------------------------------
// Kernel 3: gather selected expert weight, transpose, convert to bf16
//   W  (fp32): [NUM_OPS][N_EXPERTS][R][C] row-major
//   out(bf16): [C][R]   (i.e. B^T layout [N][K] for the GEMM)
// ------------------------------------------------------------------
__global__ __launch_bounds__(256) void moe_wtranspose_kernel(const float* __restrict__ W,
                                                             const int* __restrict__ opPtr,
                                                             int expert, int R, int C,
                                                             __bf16* __restrict__ out) {
    __shared__ float tile[32][33];
    const int op = opPtr[0];
    const float* Wp = W + ((size_t)op * N_EXPERTS + expert) * (size_t)R * (size_t)C;
    const int tx = threadIdx.x & 31;
    const int ty = threadIdx.x >> 5;          // 0..7
    const int c0 = blockIdx.x * 32;
    const int r0 = blockIdx.y * 32;
    #pragma unroll
    for (int i = 0; i < 32; i += 8)
        tile[ty + i][tx] = Wp[(size_t)(r0 + ty + i) * C + (c0 + tx)];
    __syncthreads();
    #pragma unroll
    for (int i = 0; i < 32; i += 8)
        out[(size_t)(c0 + ty + i) * R + (r0 + tx)] = (__bf16)tile[tx][ty + i];
}

// ------------------------------------------------------------------
// Kernel 4: tiled bf16 WMMA GEMM, C = A[M,K] * B^T[N,K]^T + bias (+GELU)
//   block tile 128x128x32, 8 waves, wave tile 32x64 => 8 wmma / wave / kstep
//   double-buffered LDS; async global->LDS staging when available
// ------------------------------------------------------------------
template <int ACT_GELU, int OUT_BF16>
__global__ __launch_bounds__(256) void moe_gemm_wmma_kernel(
    const __bf16* __restrict__ A,     // [M][K]
    const __bf16* __restrict__ BT,    // [N][K]
    const float*  __restrict__ biasAll,   // [NUM_OPS][N_EXPERTS][N]
    const int*    __restrict__ opPtr,
    int expert, int M, int N, int K,
    float* __restrict__ outF, __bf16* __restrict__ outB) {

    constexpr int BM = 128, BN = 128, BK = 32, PAD = 8, LROW = BK + PAD;
    __shared__ __bf16 As[2][BM][LROW];    // 2 x 10240 B
    __shared__ __bf16 Bs[2][BN][LROW];    // 2 x 10240 B

    const int t    = threadIdx.x;
    const int wid  = t >> 5;
    const int lane = t & 31;
    const int wm   = wid & 3;             // 4 waves along M
    const int wn   = wid >> 2;            // 2 waves along N
    const int half = lane >> 4;           // K-half selector (ISA A/B 16-bit layout)
    const int r    = lane & 15;           // row (A) / col (B,D) within 16-tile

    const int m0 = blockIdx.y * BM;
    const int n0 = blockIdx.x * BN;

    // chunk assignment: idx in [0,512): row = idx>>2, col8 = (idx&3)*8
    const int rA0 = (t)        >> 2, c0b = ((t)        & 3) * 8;
    const int rA1 = (t + 256)  >> 2, c1b = ((t + 256)  & 3) * 8;

    const __bf16* gA0 = A  + ((size_t)(m0 + rA0)) * K + c0b;
    const __bf16* gA1 = A  + ((size_t)(m0 + rA1)) * K + c1b;
    const __bf16* gB0 = BT + ((size_t)(n0 + rA0)) * K + c0b;
    const __bf16* gB1 = BT + ((size_t)(n0 + rA1)) * K + c1b;

    floatx8 acc[2][4] = {};               // 64 VGPRs of f32 accumulators

    auto stage = [&](int buf, size_t koff) {
#if USE_ASYNC_LDS
        async_b128(gA0 + koff, &As[buf][rA0][c0b]);
        async_b128(gA1 + koff, &As[buf][rA1][c1b]);
        async_b128(gB0 + koff, &Bs[buf][rA0][c0b]);
        async_b128(gB1 + koff, &Bs[buf][rA1][c1b]);
#else
        *(bf16x8*)&As[buf][rA0][c0b] = *(const bf16x8*)(gA0 + koff);
        *(bf16x8*)&As[buf][rA1][c1b] = *(const bf16x8*)(gA1 + koff);
        *(bf16x8*)&Bs[buf][rA0][c0b] = *(const bf16x8*)(gB0 + koff);
        *(bf16x8*)&Bs[buf][rA1][c1b] = *(const bf16x8*)(gB1 + koff);
#endif
    };

    const int nk = K / BK;
    stage(0, 0);                          // prologue: tile 0 in flight

    for (int kt = 0; kt < nk; ++kt) {
        const int buf = kt & 1;
        const bool hasNext = (kt + 1) < nk;
        if (hasNext) stage(buf ^ 1, (size_t)(kt + 1) * BK);   // keep next tile in flight

#if USE_ASYNC_LDS
        if (hasNext) __builtin_amdgcn_s_wait_asynccnt(4);     // tile kt landed, kt+1 in flight
        else         __builtin_amdgcn_s_wait_asynccnt(0);
#else
        if (hasNext) {
            __builtin_prefetch(gA0 + (size_t)(kt + 1) * BK, 0, 0);
            __builtin_prefetch(gB0 + (size_t)(kt + 1) * BK, 0, 0);
        }
#endif
        __syncthreads();                  // all waves' tile-kt copies visible

        // ---- build WMMA fragments per ISA 16-bit layouts ----
        bf16x16 af[2];
        #pragma unroll
        for (int tm = 0; tm < 2; ++tm) {
            int row = wm * 32 + tm * 16 + r;
            // lanes 0-15: K 0..7 and 16..23 ; lanes 16-31: K 8..15 and 24..31
            bf16x8 lo = *(const bf16x8*)&As[buf][row][half * 8];
            bf16x8 hi = *(const bf16x8*)&As[buf][row][16 + half * 8];
            af[tm] = __builtin_shufflevector(lo, hi, 0,1,2,3,4,5,6,7,8,9,10,11,12,13,14,15);
        }
        bf16x16 bfrag[4];
        #pragma unroll
        for (int tn = 0; tn < 4; ++tn) {
            int col = wn * 64 + tn * 16 + r;
            // lanes 0-15: K 0..7 and 16..23 ; lanes 16-31: K 8..15 and 24..31
            bf16x8 lo = *(const bf16x8*)&Bs[buf][col][half * 8];
            bf16x8 hi = *(const bf16x8*)&Bs[buf][col][16 + half * 8];
            bfrag[tn] = __builtin_shufflevector(lo, hi, 0,1,2,3,4,5,6,7,8,9,10,11,12,13,14,15);
        }

        // ---- 8 matrix MACs: D = A*B + C ----
        #pragma unroll
        for (int tm = 0; tm < 2; ++tm)
            #pragma unroll
            for (int tn = 0; tn < 4; ++tn)
                acc[tm][tn] = __builtin_amdgcn_wmma_f32_16x16x32_bf16(
                    false, af[tm], false, bfrag[tn], (short)0, acc[tm][tn], false, false);

        __syncthreads();                  // buf may be overwritten next iteration
    }

    // ---- epilogue: bias (+ tanh-GELU), store ----
    const int op = opPtr[0];
    const float* bias = biasAll + ((size_t)op * N_EXPERTS + expert) * (size_t)N;

    #pragma unroll
    for (int tm = 0; tm < 2; ++tm) {
        #pragma unroll
        for (int tn = 0; tn < 4; ++tn) {
            const int n  = n0 + wn * 64 + tn * 16 + r;        // D layout: n = lane%16
            const float bv = bias[n];
            const int mbase = m0 + wm * 32 + tm * 16 + half * 8; // m = 8*(lane/16)+q
            #pragma unroll
            for (int q = 0; q < 8; ++q) {
                float v = acc[tm][tn][q] + bv;
                if (ACT_GELU) {
                    float u = v * (0.7978845608f + 0.0356774081f * v * v);
                    v = 0.5f * v * (1.0f + tanhf(u));
                }
                size_t off = (size_t)(mbase + q) * N + n;
                if (OUT_BF16) outB[off] = (__bf16)v;
                else          outF[off] = v;
            }
        }
    }
}

// ------------------------------------------------------------------
// launch: argmax -> cvt x -> per expert { Wt, GEMM1(+gelu), Wt, GEMM2 }
// ------------------------------------------------------------------
extern "C" void kernel_launch(void* const* d_in, const int* in_sizes, int n_in,
                              void* d_out, int out_size, void* d_ws, size_t ws_size,
                              hipStream_t stream) {
    const float* x  = (const float*)d_in[0];
    const float* W1 = (const float*)d_in[1];
    const float* b1 = (const float*)d_in[2];
    const float* W2 = (const float*)d_in[3];
    const float* b2 = (const float*)d_in[4];
    float* out = (float*)d_out;

    // ws layout (bytes): [op int | x_bf16 16MB | h_bf16 64MB | wt1 8MB | wt2 8MB]
    char* ws = (char*)d_ws;
    int*    opBuf = (int*)ws;
    const size_t szX = (size_t)TOKENS * D_MODEL * 2;   // 16 MB
    const size_t szH = (size_t)TOKENS * D_FF * 2;      // 64 MB
    const size_t szW = (size_t)D_MODEL * D_FF * 2;     // 8 MB
    __bf16* xb  = (__bf16*)(ws + 256);
    __bf16* hb  = (__bf16*)(ws + 256 + szX);
    __bf16* wt1 = (__bf16*)(ws + 256 + szX + szH);
    __bf16* wt2 = (__bf16*)(ws + 256 + szX + szH + szW);

    moe_argmax_kernel<<<1, 32, 0, stream>>>(x, opBuf);

    const int n4 = (TOKENS * D_MODEL) / 4;
    moe_cvt_bf16_kernel<<<(n4 + 255) / 256, 256, 0, stream>>>(x, xb, n4);

    const dim3 tg1(D_FF / 32, D_MODEL / 32);   // W1[D][F] -> wt1[F][D]
    const dim3 tg2(D_MODEL / 32, D_FF / 32);   // W2[F][D] -> wt2[D][F]
    const dim3 g1(D_FF / 128, TOKENS / 128);   // 32 x 64 blocks
    const dim3 g2(D_MODEL / 128, TOKENS / 128);// 8 x 64 blocks

    // expert 0
    moe_wtranspose_kernel<<<tg1, 256, 0, stream>>>(W1, opBuf, 0, D_MODEL, D_FF, wt1);
    moe_gemm_wmma_kernel<1, 1><<<g1, 256, 0, stream>>>(xb, wt1, b1, opBuf, 0,
                                                       TOKENS, D_FF, D_MODEL, nullptr, hb);
    moe_wtranspose_kernel<<<tg2, 256, 0, stream>>>(W2, opBuf, 0, D_FF, D_MODEL, wt2);
    moe_gemm_wmma_kernel<0, 1><<<g2, 256, 0, stream>>>(hb, wt2, b2, opBuf, 0,
                                                       TOKENS, D_MODEL, D_FF, nullptr, xb);
    // expert 1 (reads xb written by stage above; final stage writes fp32 d_out)
    moe_wtranspose_kernel<<<tg1, 256, 0, stream>>>(W1, opBuf, 1, D_MODEL, D_FF, wt1);
    moe_gemm_wmma_kernel<1, 1><<<g1, 256, 0, stream>>>(xb, wt1, b1, opBuf, 1,
                                                       TOKENS, D_FF, D_MODEL, nullptr, hb);
    moe_wtranspose_kernel<<<tg2, 256, 0, stream>>>(W2, opBuf, 1, D_FF, D_MODEL, wt2);
    moe_gemm_wmma_kernel<0, 0><<<g2, 256, 0, stream>>>(hb, wt2, b2, opBuf, 1,
                                                       TOKENS, D_MODEL, D_FF, out, nullptr);
}
